// Attn_8804682957108
// MI455X (gfx1250) — matrix operand order repaired and checked by reading, not compiled, not run on hardware
//
#include <hip/hip_runtime.h>
#include <hip/hip_bf16.h>

// ---------------------------------------------------------------------------
// Problem constants
// ---------------------------------------------------------------------------
#define DIMC     1024
#define HEADS    16
#define HEAD_DIM 64
#define BATCH    2
#define SEQ      2048
#define MROWS    (BATCH * SEQ)      // 4096
#define LN_EPS   1e-5f

typedef __attribute__((ext_vector_type(16))) __bf16 v16bf;
typedef __attribute__((ext_vector_type(8)))  float  v8f;

union Frag16 { v16bf v; uint4 q[2]; };
union Pack2  { unsigned int u; __bf16 h[2]; };

__device__ __forceinline__ float wave_sum32(float x) {
#pragma unroll
  for (int m = 16; m >= 1; m >>= 1) x += __shfl_xor(x, m, 32);
  return x;
}

// CDNA5 async global->LDS copy (ASYNCcnt-tracked, no VGPR round trip).
// LDS byte address = low 32 bits of the generic pointer (aperture high bits
// are discarded by hardware for LDS-space addresses).
__device__ __forceinline__ void async_copy_b128(void* lds, const void* gptr) {
  unsigned l = (unsigned)(unsigned long long)lds;
  asm volatile("global_load_async_to_lds_b128 %0, %1, off"
               :: "v"(l), "v"(gptr) : "memory");
}
__device__ __forceinline__ void wait_async0() {
  asm volatile("s_wait_asynccnt 0" ::: "memory");
}

// ---------------------------------------------------------------------------
// Kernel 1a: f32 -> bf16 conversion (grid-stride)
// ---------------------------------------------------------------------------
__global__ void f32_to_bf16(const float* __restrict__ src,
                            __bf16* __restrict__ dst, int n) {
  int i = blockIdx.x * blockDim.x + threadIdx.x;
  int stride = gridDim.x * blockDim.x;
  for (; i < n; i += stride) dst[i] = (__bf16)src[i];
}

// ---------------------------------------------------------------------------
// Kernel 1b: f32 -> bf16 transposing convert: dst[c][r] = src[r][c]
//   32x32 LDS tile, coalesced on both sides. block = 256 (32x8)
// ---------------------------------------------------------------------------
__global__ __launch_bounds__(256)
void transpose_f32_bf16(const float* __restrict__ src, __bf16* __restrict__ dst,
                        int R, int Ccols) {
  __shared__ float tile[32][33];
  const int c0 = blockIdx.x * 32, r0 = blockIdx.y * 32;
  const int tx = threadIdx.x & 31, ty = threadIdx.x >> 5;
#pragma unroll
  for (int i = 0; i < 4; ++i)
    tile[ty + i * 8][tx] = src[(size_t)(r0 + ty + i * 8) * Ccols + c0 + tx];
  __syncthreads();
#pragma unroll
  for (int i = 0; i < 4; ++i)
    dst[(size_t)(c0 + ty + i * 8) * R + r0 + tx] = (__bf16)tile[tx][ty + i * 8];
}

// ---------------------------------------------------------------------------
// Kernel 2/5: bf16 WMMA GEMM  C[M,N] = A[M,K] * BT[N,K]^T  (f32 out)
//   Block tile 256x64, BK=32, 256 threads = 8 waves, wave tile 32x64
//   Double-buffered LDS fed by async global->LDS b128 copies.
// ---------------------------------------------------------------------------
#define BM 256
#define BN 64
#define BK 32

__global__ __launch_bounds__(256)
void gemm_bf16(const __bf16* __restrict__ A, const __bf16* __restrict__ BT,
               float* __restrict__ C, int M, int N, int K) {
  __shared__ __align__(16) __bf16 sA[2][BM * BK];   // [m][k], ld=32
  __shared__ __align__(16) __bf16 sBT[2][BN * BK];  // [n][k], ld=32

  const int t    = threadIdx.x;
  const int lane = t & 31;
  const int w    = t >> 5;        // 0..7 -> wave rows w*32..+31
  const int m0   = blockIdx.y * BM;
  const int n0   = blockIdx.x * BN;
  const int half = lane >> 4;
  const int l16  = lane & 15;

  const int bn = t >> 2;           // 0..63  (BT row)
  const int bk = (t & 3) * 8;      // 0,8,16,24

  v8f acc[2][4] = {};

  // async-stage one K-step: A rows = thread id (256 rows), 64B each;
  // BT: 64 rows x 64B.
  auto stage = [&](int buf, int k0) {
    const __bf16* ga = A + (size_t)(m0 + t) * K + k0;
    __bf16* la = &sA[buf][t * BK];
#pragma unroll
    for (int c = 0; c < 4; ++c) async_copy_b128(la + c * 8, ga + c * 8);
    async_copy_b128(&sBT[buf][bn * BK + bk], BT + (size_t)(n0 + bn) * K + k0 + bk);
  };

  stage(0, 0);
  wait_async0();
  __syncthreads();

  int buf = 0;
  for (int k0 = 0; k0 < K; k0 += BK) {
    if (k0 + BK < K) stage(buf ^ 1, k0 + BK);  // overlap copy with WMMA

    Frag16 bfr[4];
#pragma unroll
    for (int ni = 0; ni < 4; ++ni) {
      const __bf16* p = &sBT[buf][(ni * 16 + l16) * BK + half * 16];
      bfr[ni].q[0] = *(const uint4*)(p);
      bfr[ni].q[1] = *(const uint4*)(p + 8);
    }
    Frag16 afr[2];
#pragma unroll
    for (int mi = 0; mi < 2; ++mi) {
      const __bf16* p = &sA[buf][(w * 32 + mi * 16 + l16) * BK + half * 16];
      afr[mi].q[0] = *(const uint4*)(p);
      afr[mi].q[1] = *(const uint4*)(p + 8);
    }
#pragma unroll
    for (int mi = 0; mi < 2; ++mi)
#pragma unroll
      for (int ni = 0; ni < 4; ++ni)
        acc[mi][ni] = __builtin_amdgcn_wmma_f32_16x16x32_bf16(
            false, afr[mi].v, false, bfr[ni].v, (short)0, acc[mi][ni], false, false);

    wait_async0();      // next buffer's copies have landed
    __syncthreads();    // all waves done with 'buf', next buffer visible
    buf ^= 1;
  }

#pragma unroll
  for (int mi = 0; mi < 2; ++mi)
#pragma unroll
    for (int ni = 0; ni < 4; ++ni)
#pragma unroll
      for (int r = 0; r < 8; ++r) {
        int m = m0 + w * 32 + mi * 16 + half * 8 + r;
        int n = n0 + ni * 16 + l16;
        C[(size_t)m * N + n] = acc[mi][ni][r];
      }
}

// ---------------------------------------------------------------------------
// Kernel 3: QKV epilogue — per-head LN + RoPE on q/k, V written TRANSPOSED
//           ([b,h,d,s], so flash can async-stage V^T tiles), sigmoid(gate).
// ---------------------------------------------------------------------------
__global__ __launch_bounds__(256)
void qkv_epilogue(const float* __restrict__ qkv, const float* __restrict__ freqs,
                  const float* __restrict__ qn_w, const float* __restrict__ qn_b,
                  const float* __restrict__ kn_w, const float* __restrict__ kn_b,
                  __bf16* __restrict__ Qb, __bf16* __restrict__ Kb,
                  __bf16* __restrict__ VbT, __bf16* __restrict__ Gs) {
  const int lane = threadIdx.x & 31;
  const int row  = blockIdx.x * 8 + (threadIdx.x >> 5);  // 0 .. B*S*H-1
  const int h    = row & (HEADS - 1);
  const int s    = (row >> 4) & (SEQ - 1);
  const int b    = row >> 15;

  const float* base = qkv + (size_t)(b * SEQ + s) * (4 * DIMC);
  const int d0 = 2 * lane;

  const float fr = freqs[s * (HEAD_DIM / 2) + lane];
  const float c = __cosf(fr), sn = __sinf(fr);

  const size_t hsd = ((size_t)(b * HEADS + h) * SEQ + s) * HEAD_DIM + d0;

  // ---- Q: LN + RoPE ----
  {
    float2 v = *(const float2*)(base + h * HEAD_DIM + d0);
    float mu  = wave_sum32(v.x + v.y) * (1.0f / 64.0f);
    float d1 = v.x - mu, d2 = v.y - mu;
    float var = wave_sum32(d1 * d1 + d2 * d2) * (1.0f / 64.0f);
    float rs = rsqrtf(var + LN_EPS);
    float n1 = d1 * rs * qn_w[d0] + qn_b[d0];
    float n2 = d2 * rs * qn_w[d0 + 1] + qn_b[d0 + 1];
    Pack2 pk;
    pk.h[0] = (__bf16)(n1 * c - n2 * sn);
    pk.h[1] = (__bf16)(n1 * sn + n2 * c);
    *(unsigned int*)(Qb + hsd) = pk.u;
  }
  // ---- K: LN + RoPE ----
  {
    float2 v = *(const float2*)(base + DIMC + h * HEAD_DIM + d0);
    float mu  = wave_sum32(v.x + v.y) * (1.0f / 64.0f);
    float d1 = v.x - mu, d2 = v.y - mu;
    float var = wave_sum32(d1 * d1 + d2 * d2) * (1.0f / 64.0f);
    float rs = rsqrtf(var + LN_EPS);
    float n1 = d1 * rs * kn_w[d0] + kn_b[d0];
    float n2 = d2 * rs * kn_w[d0 + 1] + kn_b[d0 + 1];
    Pack2 pk;
    pk.h[0] = (__bf16)(n1 * c - n2 * sn);
    pk.h[1] = (__bf16)(n1 * sn + n2 * c);
    *(unsigned int*)(Kb + hsd) = pk.u;
  }
  // ---- V: bf16, transposed to [b,h,d,s] ----
  {
    float2 v = *(const float2*)(base + 2 * DIMC + h * HEAD_DIM + d0);
    __bf16* vt = VbT + (size_t)(b * HEADS + h) * HEAD_DIM * SEQ;
    vt[(size_t)d0 * SEQ + s]       = (__bf16)v.x;
    vt[(size_t)(d0 + 1) * SEQ + s] = (__bf16)v.y;
  }
  // ---- gate: sigmoid ----
  {
    float2 v = *(const float2*)(base + 3 * DIMC + h * HEAD_DIM + d0);
    Pack2 pk;
    pk.h[0] = (__bf16)(1.0f / (1.0f + __expf(-v.x)));
    pk.h[1] = (__bf16)(1.0f / (1.0f + __expf(-v.y)));
    *(unsigned int*)(Gs + (size_t)(b * SEQ + s) * DIMC + h * HEAD_DIM + d0) = pk.u;
  }
}

// ---------------------------------------------------------------------------
// Kernel 4: flash attention per (b,h): 128 queries/block, 8 waves x 16 rows,
//           key tiles of 64 async-staged into LDS. Online softmax, fused gate.
// ---------------------------------------------------------------------------
#define FA_BM 128
#define FA_BK 64

__global__ __launch_bounds__(256)
void flash_attn(const __bf16* __restrict__ Qg, const __bf16* __restrict__ Kg,
                const __bf16* __restrict__ VT, const __bf16* __restrict__ Gs,
                __bf16* __restrict__ Ob) {
  __shared__ __align__(16) __bf16 sK[FA_BK * HEAD_DIM];    // [key][d]  (B^T for Q*K^T)
  __shared__ __align__(16) __bf16 sVT[HEAD_DIM * FA_BK];   // [d][key]  (B^T for P*V)
  __shared__ __align__(16) __bf16 sP[8 * 16 * HEAD_DIM];   // wave-private P tiles

  const int t    = threadIdx.x;
  const int lane = t & 31;
  const int w    = t >> 5;
  const int half = lane >> 4;
  const int l16  = lane & 15;
  const int bh   = blockIdx.y;        // 0..31
  const int b    = bh >> 4;
  const int h    = bh & 15;
  const int q0   = blockIdx.x * FA_BM + w * 16;

  const size_t headBase = (size_t)bh * SEQ * HEAD_DIM;
  const float scale = 0.125f;  // 1/sqrt(64)

  // Q fragments for this wave's 16 rows (reused across all key tiles)
  Frag16 qf[2];
#pragma unroll
  for (int di = 0; di < 2; ++di) {
    const __bf16* p = Qg + headBase + (size_t)(q0 + l16) * HEAD_DIM + di * 32 + half * 16;
    qf[di].q[0] = *(const uint4*)(p);
    qf[di].q[1] = *(const uint4*)(p + 8);
  }

  v8f oacc[4] = {};
  float mrow[8], lrow[8];
#pragma unroll
  for (int r = 0; r < 8; ++r) { mrow[r] = -1e30f; lrow[r] = 0.0f; }

  __bf16* sPw = sP + w * (16 * HEAD_DIM);
  const int krow = t >> 2;         // 0..63
  const int kcol = (t & 3) * 16;   // 0,16,32,48

  for (int kb = 0; kb < SEQ; kb += FA_BK) {
    // async-stage K tile [key][d] and V^T tile [d][key]
    {
      const __bf16* gk = Kg + headBase + (size_t)(kb + krow) * HEAD_DIM + kcol;
      async_copy_b128(sK + krow * HEAD_DIM + kcol,     gk);
      async_copy_b128(sK + krow * HEAD_DIM + kcol + 8, gk + 8);
      const __bf16* gv = VT + ((size_t)bh * HEAD_DIM + krow) * SEQ + kb + kcol;
      async_copy_b128(sVT + krow * FA_BK + kcol,     gv);
      async_copy_b128(sVT + krow * FA_BK + kcol + 8, gv + 8);
    }
    wait_async0();
    __syncthreads();

    // ---- scores: S(16x64) = Q(16x64) * K^T ----
    v8f sc[4] = {};
#pragma unroll
    for (int di = 0; di < 2; ++di)
#pragma unroll
      for (int ni = 0; ni < 4; ++ni) {
        Frag16 kf;
        const __bf16* p = sK + (ni * 16 + l16) * HEAD_DIM + di * 32 + half * 16;
        kf.q[0] = *(const uint4*)(p);
        kf.q[1] = *(const uint4*)(p + 8);
        sc[ni] = __builtin_amdgcn_wmma_f32_16x16x32_bf16(
            false, qf[di].v, false, kf.v, (short)0, sc[ni], false, false);
      }

#pragma unroll
    for (int ni = 0; ni < 4; ++ni)
#pragma unroll
      for (int r = 0; r < 8; ++r) sc[ni][r] *= scale;

    // ---- online softmax (rows half*8+r live in this lane half) ----
    float alpha[8];
#pragma unroll
    for (int r = 0; r < 8; ++r) {
      float mx = fmaxf(fmaxf(sc[0][r], sc[1][r]), fmaxf(sc[2][r], sc[3][r]));
#pragma unroll
      for (int msk = 8; msk >= 1; msk >>= 1) mx = fmaxf(mx, __shfl_xor(mx, msk, 32));
      float mnew = fmaxf(mrow[r], mx);
      alpha[r] = __expf(mrow[r] - mnew);
      mrow[r] = mnew;
    }
#pragma unroll
    for (int r = 0; r < 8; ++r) {
      float rsum = 0.0f;
#pragma unroll
      for (int ni = 0; ni < 4; ++ni) {
        float pv = __expf(sc[ni][r] - mrow[r]);
        rsum += pv;
        sPw[(half * 8 + r) * HEAD_DIM + ni * 16 + l16] = (__bf16)pv;  // C->A via LDS
      }
#pragma unroll
      for (int msk = 8; msk >= 1; msk >>= 1) rsum += __shfl_xor(rsum, msk, 32);
      lrow[r] = lrow[r] * alpha[r] + rsum;
#pragma unroll
      for (int ni = 0; ni < 4; ++ni) oacc[ni][r] *= alpha[r];
    }
    asm volatile("s_wait_dscnt 0" ::: "memory");  // wave-private LDS RAW fence

    // ---- O += P(16x64) * V(64x64) ----
#pragma unroll
    for (int ks = 0; ks < 2; ++ks) {
      Frag16 pf;
      const __bf16* pp = sPw + l16 * HEAD_DIM + ks * 32 + half * 16;
      pf.q[0] = *(const uint4*)(pp);
      pf.q[1] = *(const uint4*)(pp + 8);
#pragma unroll
      for (int ni = 0; ni < 4; ++ni) {
        Frag16 vf;
        const __bf16* vp = sVT + (ni * 16 + l16) * FA_BK + ks * 32 + half * 16;
        vf.q[0] = *(const uint4*)(vp);
        vf.q[1] = *(const uint4*)(vp + 8);
        oacc[ni] = __builtin_amdgcn_wmma_f32_16x16x32_bf16(
            false, pf.v, false, vf.v, (short)0, oacc[ni], false, false);
      }
    }
    __syncthreads();
  }

  // ---- finalize: 1/l normalize, sigmoid-gate, store bf16 ----
#pragma unroll
  for (int r = 0; r < 8; ++r) {
    float inv = 1.0f / lrow[r];
    int s = q0 + half * 8 + r;
    size_t gbase = (size_t)(b * SEQ + s) * DIMC + h * HEAD_DIM;
#pragma unroll
    for (int ni = 0; ni < 4; ++ni) {
      int d = ni * 16 + l16;
      float g = (float)Gs[gbase + d];
      Ob[gbase + d] = (__bf16)(oacc[ni][r] * inv * g);
    }
  }
}

// ---------------------------------------------------------------------------
// Host-side launcher
// ---------------------------------------------------------------------------
extern "C" void kernel_launch(void* const* d_in, const int* in_sizes, int n_in,
                              void* d_out, int out_size, void* d_ws, size_t ws_size,
                              hipStream_t stream) {
  const float* x     = (const float*)d_in[0];
  const float* freqs = (const float*)d_in[1];
  const float* w_qkv = (const float*)d_in[2];
  const float* w_out = (const float*)d_in[3];
  const float* qn_w  = (const float*)d_in[4];
  const float* qn_b  = (const float*)d_in[5];
  const float* kn_w  = (const float*)d_in[6];
  const float* kn_b  = (const float*)d_in[7];
  float* out = (float*)d_out;

  char* ws = (char*)d_ws;
  __bf16* xb     = (__bf16*)(ws + (size_t)0);            //  8 MiB  x bf16
  __bf16* wqkvT  = (__bf16*)(ws + ((size_t)8  << 20));   //  8 MiB  w_qkv^T bf16 [4096][1024]
  __bf16* woutT  = (__bf16*)(ws + ((size_t)16 << 20));   //  2 MiB  w_out^T bf16 [1024][1024]
  float*  qkv    = (float* )(ws + ((size_t)18 << 20));   // 64 MiB  qkv f32
  __bf16* Qb     = (__bf16*)(ws + ((size_t)82 << 20));   //  8 MiB  [b,h,s,d]
  __bf16* Kb     = (__bf16*)(ws + ((size_t)90 << 20));   //  8 MiB  [b,h,s,d]
  __bf16* VbT    = (__bf16*)(ws + ((size_t)98 << 20));   //  8 MiB  [b,h,d,s]
  __bf16* Gs     = (__bf16*)(ws + ((size_t)106 << 20));  //  8 MiB  sigmoid(gate)
  __bf16* Ob     = (__bf16*)(ws + ((size_t)114 << 20));  //  8 MiB  gated attn out

  f32_to_bf16<<<dim3(4096), dim3(256), 0, stream>>>(x, xb, MROWS * DIMC);
  // weights converted + transposed once (N-major) so GEMM B-tiles are contiguous
  transpose_f32_bf16<<<dim3(4 * DIMC / 32, DIMC / 32), dim3(256), 0, stream>>>(
      w_qkv, wqkvT, DIMC, 4 * DIMC);
  transpose_f32_bf16<<<dim3(DIMC / 32, DIMC / 32), dim3(256), 0, stream>>>(
      w_out, woutT, DIMC, DIMC);

  // qkv = x @ w_qkv  (4096 x 4096, K=1024)
  gemm_bf16<<<dim3(4 * DIMC / BN, MROWS / BM), dim3(256), 0, stream>>>(
      xb, wqkvT, qkv, MROWS, 4 * DIMC, DIMC);

  qkv_epilogue<<<dim3(MROWS * HEADS / 8), dim3(256), 0, stream>>>(
      qkv, freqs, qn_w, qn_b, kn_w, kn_b, Qb, Kb, VbT, Gs);

  flash_attn<<<dim3(SEQ / FA_BM, BATCH * HEADS), dim3(256), 0, stream>>>(
      Qb, Kb, VbT, Gs, Ob);

  // out = Ob @ w_out  (4096 x 1024, K=1024)
  gemm_bf16<<<dim3(DIMC / BN, MROWS / BM), dim3(256), 0, stream>>>(
      Ob, woutT, out, MROWS, DIMC, DIMC);
}
